// GCNNet1_57243324121150
// MI455X (gfx1250) — matrix operand order repaired and checked by reading, not compiled, not run on hardware
//
#include <hip/hip_runtime.h>
#include <hip/hip_bf16.h>

typedef __attribute__((ext_vector_type(16))) _Float16 v16h;
typedef __attribute__((ext_vector_type(8)))  _Float16 v8h;
typedef __attribute__((ext_vector_type(8)))  float    v8f;

#define GCN_HID 146
#define GCN_G   256
#define GCN_EPS 1e-5f

// ---------------------------------------------------------------------------
// Zero a float buffer (graph-capture-safe memset)
// ---------------------------------------------------------------------------
__global__ void zero_kernel(float* __restrict__ p, long long n) {
    long long i = (long long)blockIdx.x * blockDim.x + threadIdx.x;
    long long stride = (long long)gridDim.x * blockDim.x;
    for (; i < n; i += stride) p[i] = 0.0f;
}

// ---------------------------------------------------------------------------
// Degrees + normalization factors
// ---------------------------------------------------------------------------
__global__ void degree_kernel(const int* __restrict__ src, const int* __restrict__ dst,
                              float* __restrict__ dout, float* __restrict__ din, int E) {
    int i = blockIdx.x * blockDim.x + threadIdx.x;
    int stride = gridDim.x * blockDim.x;
    for (; i < E; i += stride) {
        atomicAdd(&dout[src[i]], 1.0f);
        atomicAdd(&din[dst[i]], 1.0f);
    }
}

__global__ void invsqrt_kernel(float* __restrict__ a, float* __restrict__ b, int N) {
    int i = blockIdx.x * blockDim.x + threadIdx.x;
    int stride = gridDim.x * blockDim.x;
    for (; i < N; i += stride) {
        a[i] = rsqrtf(fmaxf(a[i], 1.0f));
        b[i] = rsqrtf(fmaxf(b[i], 1.0f));
    }
}

// ---------------------------------------------------------------------------
// Edge scatter: agg[dst, f] += h[src, f] * ns[src]
// block = edge (uniform src/dst -> scalar loads), thread = feature
// ---------------------------------------------------------------------------
__global__ void edge_scatter_kernel(const float* __restrict__ h, const float* __restrict__ ns,
                                    const int* __restrict__ src, const int* __restrict__ dst,
                                    float* __restrict__ agg, int E, int C) {
    int f = threadIdx.x;
    for (int e = blockIdx.x; e < E; e += gridDim.x) {
        int s = src[e];
        int d = dst[e];
        if (f < C) {
            atomicAdd(&agg[(long long)d * C + f], h[(long long)s * C + f] * ns[s]);
        }
    }
}

// ---------------------------------------------------------------------------
// Pre-convert weights: Wt[c, k] = f16(W[k, c]), zero-padded to [Cpad x Kpad].
// ---------------------------------------------------------------------------
__global__ void convw_kernel(const float* __restrict__ W, _Float16* __restrict__ Wt,
                             int K, int C, int Kpad) {
    int c = blockIdx.x;
    int k = threadIdx.x;
    float v = (c < C && k < K) ? W[(long long)k * C + c] : 0.0f;
    Wt[(long long)c * Kpad + k] = (_Float16)v;
}

// ---------------------------------------------------------------------------
// Pre-convert activations: Af16[m, k] = f16(A[m, k] * rs[m]), zero-padded.
// ---------------------------------------------------------------------------
__global__ void conva_kernel(const float* __restrict__ A, const float* __restrict__ rs,
                             _Float16* __restrict__ Af16, int M, int K, int Kpad) {
    int m = blockIdx.x;
    int k = threadIdx.x;
    float v = 0.0f;
    if (m < M && k < K) {
        v = A[(long long)m * K + k];
        if (rs) v *= rs[m];
    }
    Af16[(long long)m * Kpad + k] = (_Float16)v;
}

// ---------------------------------------------------------------------------
// WMMA GEMM, f16 inputs / f32 accum.
//   Out[m,n] = (sum_k Af16[m,k]*Wt[n,k] + bias[n]) * (RSOUT ? rsOut[m] : 1)
// One wave owns a 16-row stripe: preloads all KPAD/32 A-fragments + the
// row-scale fragment into registers, then sweeps column tiles loading only B.
// No predication in the MMA loop; single exec mask on the fast store path.
// ---------------------------------------------------------------------------
template <int KPAD, bool RSOUT>
__global__ void wmma_gemm_kernel(const _Float16* __restrict__ Af16,
                                 const _Float16* __restrict__ Wt,
                                 const float* __restrict__ bias,
                                 const float* __restrict__ rsOut,
                                 float* __restrict__ Out,
                                 int M, int C, int nTilesM, int nTilesN) {
    constexpr int NCH = KPAD / 32;
    int wave = blockIdx.x * (blockDim.x >> 5) + (threadIdx.x >> 5);
    if (wave >= nTilesM) return;               // wave-uniform exit
    int lane = threadIdx.x & 31;
    int half = lane >> 4;                      // 0: lanes 0-15, 1: lanes 16-31
    int l16  = lane & 15;
    int row  = wave * 16 + l16;
    int mbase = wave * 16 + half * 8;          // first output row for this lane's acc

    // Preload A fragments for all k-chunks (row fixed per lane).
    v16h afrag[NCH];
#pragma unroll
    for (int cidx = 0; cidx < NCH; ++cidx) {
        const _Float16* ap = Af16 + (long long)row * KPAD + cidx * 32 + half * 16;
        v8h lo = *(const v8h*)ap;              // k = kb + half*16 + 0..7
        v8h hi = *(const v8h*)(ap + 8);        // k = kb + half*16 + 8..15
        afrag[cidx] = __builtin_shufflevector(lo, hi,
            0, 1, 2, 3, 4, 5, 6, 7, 8, 9, 10, 11, 12, 13, 14, 15);
    }

    // Preload row-scale fragment once (clamped index -> unconditional loads).
    float rsv[8];
#pragma unroll
    for (int j = 0; j < 8; ++j) {
        int m = mbase + j;
        rsv[j] = RSOUT ? rsOut[m < M ? m : (M - 1)] : 1.0f;
    }

    bool fullTile = (wave * 16 + 16 <= M);     // wave-uniform

    for (int tn = 0; tn < nTilesN; ++tn) {
        int col = tn * 16 + l16;
        bool cok = (col < C);
        v8f acc = {};
#pragma unroll
        for (int cidx = 0; cidx < NCH; ++cidx) {
            // B fragment: 16 contiguous halves at k = kb + half*16
            const _Float16* bp = Wt + (long long)col * KPAD + cidx * 32 + half * 16;
            v16h b = *(const v16h*)bp;
            acc = __builtin_amdgcn_wmma_f32_16x16x32_f16(
                      false, afrag[cidx], false, b, (short)0, acc, false, false);
        }
        float bv = bias[cok ? col : 0];        // unconditional clamped load
        if (fullTile) {
            if (cok) {
                float* op = Out + (long long)mbase * C + col;
#pragma unroll
                for (int j = 0; j < 8; ++j) {
                    op[(long long)j * C] = (acc[j] + bv) * rsv[j];
                }
            }
        } else {
#pragma unroll
            for (int j = 0; j < 8; ++j) {
                int m = mbase + j;
                if (cok && m < M) {
                    Out[(long long)m * C + col] = (acc[j] + bv) * rsv[j];
                }
            }
        }
    }
}

// ---------------------------------------------------------------------------
// BatchNorm column statistics
// ---------------------------------------------------------------------------
__global__ void colstats_kernel(const float* __restrict__ h2,
                                float* __restrict__ csum, float* __restrict__ csq,
                                int M, int C) {
    __shared__ float ssum[256];
    __shared__ float ssq[256];
    int col = blockIdx.x;
    float s = 0.0f, q = 0.0f;
    for (int m = blockIdx.y * blockDim.x + threadIdx.x; m < M;
         m += gridDim.y * blockDim.x) {
        float v = h2[(long long)m * C + col];
        s += v;
        q += v * v;
    }
    ssum[threadIdx.x] = s;
    ssq[threadIdx.x]  = q;
    __syncthreads();
    for (int off = blockDim.x >> 1; off > 0; off >>= 1) {
        if (threadIdx.x < off) {
            ssum[threadIdx.x] += ssum[threadIdx.x + off];
            ssq[threadIdx.x]  += ssq[threadIdx.x + off];
        }
        __syncthreads();
    }
    if (threadIdx.x == 0) {
        atomicAdd(&csum[col], ssum[0]);
        atomicAdd(&csq[col],  ssq[0]);
    }
}

__global__ void bnfin_kernel(const float* __restrict__ csum, const float* __restrict__ csq,
                             const float* __restrict__ gamma, const float* __restrict__ beta,
                             float* __restrict__ scale, float* __restrict__ shift,
                             int M, int C) {
    int c = blockIdx.x * blockDim.x + threadIdx.x;
    if (c < C) {
        float invM = 1.0f / (float)M;
        float mu  = csum[c] * invM;
        float var = csq[c] * invM - mu * mu;
        float sc  = gamma[c] * rsqrtf(var + GCN_EPS);
        scale[c] = sc;
        shift[c] = beta[c] - mu * sc;
    }
}

// Residual + ReLU: h[n,f] += relu(h2[n,f]*scale[f] + shift[f])
__global__ void resrelu_kernel(float* __restrict__ h, const float* __restrict__ h2,
                               const float* __restrict__ scale, const float* __restrict__ shift,
                               int N, int C) {
    int f = threadIdx.x;
    for (int n = blockIdx.x; n < N; n += gridDim.x) {
        if (f < C) {
            long long i = (long long)n * C + f;
            float v = h2[i] * scale[f] + shift[f];
            h[i] += (v > 0.0f) ? v : 0.0f;
        }
    }
}

// ---------------------------------------------------------------------------
// Readout
// ---------------------------------------------------------------------------
__global__ void counts_kernel(const int* __restrict__ gid, float* __restrict__ counts, int N) {
    int i = blockIdx.x * blockDim.x + threadIdx.x;
    int stride = gridDim.x * blockDim.x;
    for (; i < N; i += stride) atomicAdd(&counts[gid[i]], 1.0f);
}

__global__ void pool_kernel(const float* __restrict__ h, const int* __restrict__ gid,
                            float* __restrict__ out, int N, int C) {
    int f = threadIdx.x;
    for (int n = blockIdx.x; n < N; n += gridDim.x) {
        int g = gid[n];
        if (f < C) {
            atomicAdd(&out[(long long)g * C + f], h[(long long)n * C + f]);
        }
    }
}

__global__ void divide_kernel(float* __restrict__ out, const float* __restrict__ counts,
                              int total, int C) {
    int i = blockIdx.x * blockDim.x + threadIdx.x;
    if (i < total) {
        int g = i / C;
        out[i] /= fmaxf(counts[g], 1.0f);
    }
}

// ---------------------------------------------------------------------------
// Host-side orchestration
// ---------------------------------------------------------------------------
static inline int gblocks(long long n, int bs, long long cap = 1048576) {
    long long b = (n + bs - 1) / bs;
    if (b > cap) b = cap;
    if (b < 1) b = 1;
    return (int)b;
}

extern "C" void kernel_launch(void* const* d_in, const int* in_sizes, int n_in,
                              void* d_out, int out_size, void* d_ws, size_t ws_size,
                              hipStream_t stream) {
    const float* nodes = (const float*)d_in[0];
    const float* snorm = (const float*)d_in[1];
    const float* We    = (const float*)d_in[2];
    const float* be    = (const float*)d_in[3];
    const float* Wl[3]  = {(const float*)d_in[4], (const float*)d_in[8],  (const float*)d_in[12]};
    const float* bl[3]  = {(const float*)d_in[5], (const float*)d_in[9],  (const float*)d_in[13]};
    const float* gl[3]  = {(const float*)d_in[6], (const float*)d_in[10], (const float*)d_in[14]};
    const float* btl[3] = {(const float*)d_in[7], (const float*)d_in[11], (const float*)d_in[15]};
    const int* src = (const int*)d_in[16];
    const int* dst = (const int*)d_in[17];
    const int* gid = (const int*)d_in[18];

    const int N = in_sizes[1];             // snorm_n has N elements
    const int E = in_sizes[16];
    const int IN_DIM = in_sizes[0] / N;    // 64
    const int C = GCN_HID;                 // 146
    const int KP_ENC = 64;                 // padded K for encoder
    const int KP_HID = 160;                // padded K for hidden layers (146 -> 160)
    const int CPAD   = 160;                // padded column count for Wt
    float* out = (float*)d_out;

    const int nTilesM = (N + 15) / 16;
    const int nTilesN = (C + 15) / 16;     // 10
    const int Mpad = nTilesM * 16;

    // -------- workspace layout --------
    const long long nh = (long long)N * C;
    float* w      = (float*)d_ws;
    float* h      = w;                  // [N, C]
    float* agg    = h   + nh;           // [N, C]
    float* h2     = agg + nh;           // [N, C]
    float* ns     = h2  + nh;           // [N]
    float* nd     = ns  + N;            // [N]
    float* csum   = nd  + N;            // [256]
    float* csq    = csum + 256;
    float* scale  = csq  + 256;
    float* shift  = scale + 256;
    float* counts = shift + 256;        // [256]
    _Float16* Af16  = (_Float16*)(counts + 256);        // [Mpad x 160]
    _Float16* WtEnc = Af16 + (long long)Mpad * KP_HID;  // [160 x 64]
    _Float16* WtL0  = WtEnc + (long long)CPAD * KP_ENC; // [160 x 160] x3
    _Float16* WtL1  = WtL0 + (long long)CPAD * KP_HID;
    _Float16* WtL2  = WtL1 + (long long)CPAD * KP_HID;
    _Float16* WtLayer[3] = {WtL0, WtL1, WtL2};

    const int BS = 256;
    const int gemmBlocks = (nTilesM + 7) / 8;  // 8 wave32 per 256-thread block

    // -------- degrees & norms --------
    zero_kernel<<<gblocks(2LL * N, BS), BS, 0, stream>>>(ns, 2LL * N);
    degree_kernel<<<gblocks(E, BS), BS, 0, stream>>>(src, dst, ns, nd, E);
    invsqrt_kernel<<<gblocks(N, BS), BS, 0, stream>>>(ns, nd, N);

    // -------- weight pre-conversion (tiny, once per launch) --------
    convw_kernel<<<CPAD, KP_ENC, 0, stream>>>(We, WtEnc, IN_DIM, C, KP_ENC);
    for (int l = 0; l < 3; ++l)
        convw_kernel<<<CPAD, KP_HID, 0, stream>>>(Wl[l], WtLayer[l], C, C, KP_HID);

    // -------- encoder: h = nodes @ W_e + b_e --------
    conva_kernel<<<Mpad, KP_ENC, 0, stream>>>(nodes, nullptr, Af16, N, IN_DIM, KP_ENC);
    wmma_gemm_kernel<KP_ENC, false><<<gemmBlocks, BS, 0, stream>>>(
        Af16, WtEnc, be, nullptr, h, N, C, nTilesM, nTilesN);

    // -------- 3 GCN layers --------
    for (int l = 0; l < 3; ++l) {
        zero_kernel<<<gblocks(nh, BS), BS, 0, stream>>>(agg, nh);
        edge_scatter_kernel<<<gblocks(E, 1, 1048576), 160, 0, stream>>>(
            h, ns, src, dst, agg, E, C);
        // A = f16(agg * nd), zero-padded
        conva_kernel<<<Mpad, KP_HID, 0, stream>>>(agg, nd, Af16, N, C, KP_HID);
        // h2 = (A @ W + b) * snorm
        wmma_gemm_kernel<KP_HID, true><<<gemmBlocks, BS, 0, stream>>>(
            Af16, WtLayer[l], bl[l], snorm, h2, N, C, nTilesM, nTilesN);
        // batch-norm stats
        zero_kernel<<<1, BS, 0, stream>>>(csum, 512);   // csum+csq contiguous
        dim3 sgrid(C, 32);
        colstats_kernel<<<sgrid, BS, 0, stream>>>(h2, csum, csq, N, C);
        bnfin_kernel<<<1, BS, 0, stream>>>(csum, csq, gl[l], btl[l], scale, shift, N, C);
        // residual + relu
        resrelu_kernel<<<gblocks(N, 1, 1048576), 160, 0, stream>>>(h, h2, scale, shift, N, C);
    }

    // -------- graph mean readout --------
    zero_kernel<<<1, BS, 0, stream>>>(counts, GCN_G);
    zero_kernel<<<gblocks(out_size, BS), BS, 0, stream>>>(out, out_size);
    counts_kernel<<<gblocks(N, BS), BS, 0, stream>>>(gid, counts, N);
    pool_kernel<<<gblocks(N, 1, 1048576), 160, 0, stream>>>(h, gid, out, N, C);
    divide_kernel<<<gblocks(out_size, BS), BS, 0, stream>>>(out, counts, out_size, C);
}
